// EdgeUpdateLayer_31224412242359
// MI455X (gfx1250) — matrix operand order repaired and checked by reading, not compiled, not run on hardware
//
#include <hip/hip_runtime.h>
#include <hip/hip_bf16.h>

// ---------------------------------------------------------------------------
// EdgeUpdateLayer for MI455X (gfx1250, wave32, WMMA, 320KB LDS/WGP).
// bf16 WMMA GEMMs, f32 accumulate. Weights staged to LDS per block via
// global_load_async_to_lds_b128; activations gathered to LDS; GEMM1 keeps all
// 16 N-tiles in registers (single pass over A) so the H tile reuses the
// wave-private A region without barriers. LayerNorm via wave32 shfl_xor.
// ---------------------------------------------------------------------------

#define NODE_DIM 128
#define EDGE_DIM 64
#define HIDDEN   256
#define IN_DIM   320          // 2*NODE_DIM + EDGE_DIM
#define A_STRIDE 328          // IN_DIM + 8 pad (bf16 elems), multiple of 8
#define H_STRIDE 264          // HIDDEN + 8 pad (bf16 elems), multiple of 8

// LDS layout (ushort units): [idx 512][W1 81920][W2 16384][A 41984] = 275KB
#define LDS_W1_OFF   512
#define LDS_W2_OFF   (LDS_W1_OFF + 160 * 512)
#define LDS_A_OFF    (LDS_W2_OFF + 32 * 512)
#define LDS_TOTAL_US (LDS_A_OFF + 128 * A_STRIDE)

typedef __attribute__((ext_vector_type(16))) __bf16 v16bf;
typedef __attribute__((ext_vector_type(8)))  float  v8f;
typedef __attribute__((ext_vector_type(4)))  int    v4i_t;

union Frag {
    uint4 q[2];   // 32 bytes
    v16bf v;
};

__device__ __forceinline__ unsigned short f32_to_bf16_rne(float f) {
    unsigned int u = __float_as_uint(f);
    u += 0x7FFFu + ((u >> 16) & 1u);      // round-to-nearest-even
    return (unsigned short)(u >> 16);
}

// ---- async global->LDS copy (CDNA5 path) with portable fallback -----------
#if __has_builtin(__builtin_amdgcn_global_load_async_to_lds_b128) && \
    __has_builtin(__builtin_amdgcn_s_wait_asynccnt)
#define HAVE_ASYNC_LDS 1
#else
#define HAVE_ASYNC_LDS 0
#endif

__device__ __forceinline__ void copy16_g2l(const unsigned short* gsrc,
                                           unsigned short* ldst) {
#if HAVE_ASYNC_LDS
    __builtin_amdgcn_global_load_async_to_lds_b128(
        (__attribute__((address_space(1))) v4i_t*)(gsrc),
        (__attribute__((address_space(3))) v4i_t*)(ldst), 0, 0);
#else
    *(uint4*)ldst = *(const uint4*)gsrc;
#endif
}

__device__ __forceinline__ void wait_async_lds() {
#if HAVE_ASYNC_LDS
    __builtin_amdgcn_s_wait_asynccnt(0);
#endif
}

// ---------------------------------------------------------------------------
// Pack w1 [320,256] and w2 [256,64] ([K][N] f32) into bf16 WMMA B-fragments
// in d_ws. Fragment = 32(K) x 16(N); lane L holds column N = L%16 with
// K = kTile*32 + (L>=16 ? 16 : 0) + h, h = 0..15, contiguous 32B per lane.
// ---------------------------------------------------------------------------
__global__ __launch_bounds__(256) void pack_weights_kernel(
        const float* __restrict__ w1, const float* __restrict__ w2,
        unsigned short* __restrict__ wf) {
    int g = blockIdx.x * 256 + threadIdx.x;
    if (g >= 192 * 32) return;
    int lane  = g & 31;
    int frag  = g >> 5;
    int nsub  = lane & 15;
    int khalf = (lane >> 4) << 4;           // 0 or 16
    if (frag < 160) {
        int kt = frag >> 4, nt = frag & 15;
        int n  = nt * 16 + nsub;
        int kb = kt * 32 + khalf;
        unsigned short* o = wf + (size_t)frag * 512 + lane * 16;
#pragma unroll
        for (int h = 0; h < 16; ++h)
            o[h] = f32_to_bf16_rne(w1[(size_t)(kb + h) * HIDDEN + n]);
    } else {
        int f2 = frag - 160;
        int kt = f2 >> 2, nt = f2 & 3;
        int n  = nt * 16 + nsub;
        int kb = kt * 32 + khalf;
        unsigned short* o = wf + (size_t)frag * 512 + lane * 16;
#pragma unroll
        for (int h = 0; h < 16; ++h)
            o[h] = f32_to_bf16_rne(w2[(size_t)(kb + h) * EDGE_DIM + n]);
    }
}

// ---------------------------------------------------------------------------
// Main fused kernel. 256 threads = 8 waves; wave w owns 16 edges.
//   1. async-stage w1+w2 frags -> LDS; cache edge indices;  barrier
//   2. gather/concat/bf16 inputs -> LDS A-buffer; wait async; barrier
//   3. GEMM1 (all 16 N-tiles in regs, one pass over A) + bias + ReLU
//      -> H tile written into the wave's own (now dead) A region
//   4. GEMM2 + bias + residual + LayerNorm -> out   (no more barriers)
// ---------------------------------------------------------------------------
__global__ __launch_bounds__(256) void edge_update_kernel(
        const float* __restrict__ x,
        const long long* __restrict__ eidx,
        const float* __restrict__ edge_attr,
        const float* __restrict__ b1,
        const float* __restrict__ b2,
        const float* __restrict__ ln_w,
        const float* __restrict__ ln_b,
        const unsigned short* __restrict__ wf,
        float* __restrict__ out,
        int E) {
    extern __shared__ unsigned char smem_raw[];
    int* lsrc = (int*)smem_raw;                                   // [128]
    int* ldst = lsrc + 128;                                       // [128]
    unsigned short* sBase = (unsigned short*)smem_raw;
    unsigned short* sW1 = sBase + LDS_W1_OFF;                     // 160 frags
    unsigned short* sW2 = sBase + LDS_W2_OFF;                     // 32 frags
    unsigned short* sA  = sBase + LDS_A_OFF;                      // [128][A_STRIDE]

    const int tid   = threadIdx.x;
    const int lane  = tid & 31;
    const int wave  = tid >> 5;
    const int ebase = blockIdx.x * 128;

    // ---- Phase 1: async-stage all weight fragments (192KB) into LDS -------
#pragma unroll 4
    for (int c = tid; c < 160 * 512 / 8; c += 256)
        copy16_g2l(wf + (size_t)c * 8, sW1 + (size_t)c * 8);
#pragma unroll 2
    for (int c = tid; c < 32 * 512 / 8; c += 256)
        copy16_g2l(wf + (size_t)(160 * 512) + (size_t)c * 8, sW2 + (size_t)c * 8);

    if (tid < 128) {
        int ge = ebase + tid;
        long long s = 0, d = 0;
        if (ge < E) { s = eidx[ge]; d = eidx[(size_t)E + ge]; }
        lsrc[tid] = (int)s;
        ldst[tid] = (int)d;
    }
    __syncthreads();

    // ---- Phase 2: gather [x_i | x_j | edge_attr] rows into LDS as bf16 ----
    for (int i = tid; i < 128 * (IN_DIM / 2); i += 256) {
        int row = i / (IN_DIM / 2);
        int col = (i - row * (IN_DIM / 2)) * 2;
        int ge  = ebase + row;
        float2 v = make_float2(0.f, 0.f);
        if (ge < E) {
            const float* p;
            if (col < NODE_DIM)
                p = x + (size_t)ldst[row] * NODE_DIM + col;              // x_i
            else if (col < 2 * NODE_DIM)
                p = x + (size_t)lsrc[row] * NODE_DIM + (col - NODE_DIM); // x_j
            else
                p = edge_attr + (size_t)ge * EDGE_DIM + (col - 2 * NODE_DIM);
            v = *(const float2*)p;
        }
        unsigned int packed = (unsigned int)f32_to_bf16_rne(v.x)
                            | ((unsigned int)f32_to_bf16_rne(v.y) << 16);
        *(unsigned int*)(sA + (size_t)row * A_STRIDE + col) = packed;
    }
    wait_async_lds();          // our async weight chunks landed
    __syncthreads();           // all weights + A-buffer visible

    const int tilebase = ebase + wave * 16;
    if (tilebase >= E) return;          // no barriers remain below

    const int m  = lane & 15;    // A-frag row / C-frag column
    const int hi = lane >> 4;    // lane half: K sub-range select / M+8
    unsigned short* sAreg = sA + (size_t)(wave * 16) * A_STRIDE;  // wave-private
    const unsigned short* arow = sAreg + (size_t)m * A_STRIDE + hi * 16;

    // ---- Phase 3: GEMM1 [16,320]x[320,256], all 16 N-tiles resident --------
    v8f acc[16];
#pragma unroll
    for (int nt = 0; nt < 16; ++nt) {
        v8f z = {0.f, 0.f, 0.f, 0.f, 0.f, 0.f, 0.f, 0.f};
        acc[nt] = z;
    }
#pragma unroll
    for (int kt = 0; kt < 10; ++kt) {
        Frag a;
        const unsigned short* ap = arow + kt * 32;
        a.q[0] = *(const uint4*)(ap);        // halfs 0..7
        a.q[1] = *(const uint4*)(ap + 8);    // halfs 8..15
#pragma unroll
        for (int nt = 0; nt < 16; ++nt) {
            const unsigned short* bp = sW1 +
                ((size_t)(kt * 16 + nt) * 512 + lane * 16);
            Frag b;
            b.q[0] = *(const uint4*)(bp);
            b.q[1] = *(const uint4*)(bp + 8);
            acc[nt] = __builtin_amdgcn_wmma_f32_16x16x32_bf16(
                    false, a.v, false, b.v, (short)0, acc[nt], false, false);
        }
    }

    // Bias + ReLU + bf16; H tile overwrites the wave's own dead A region.
    // C layout: VGPR r, lane L -> row = r + hi*8, col = nt*16 + m.
#pragma unroll
    for (int nt = 0; nt < 16; ++nt) {
        int col = nt * 16 + m;
        float bv = b1[col];
#pragma unroll
        for (int r = 0; r < 8; ++r) {
            int row = r + hi * 8;
            float hv = acc[nt][r] + bv;
            hv = hv > 0.f ? hv : 0.f;
            sAreg[(size_t)row * H_STRIDE + col] = f32_to_bf16_rne(hv);
        }
    }
    asm volatile("" ::: "memory");  // per-wave LDS ops retire in order (DScnt)

    // ---- Phase 4: GEMM2 [16,256]x[256,64] ---------------------------------
    v8f acc2[4];
#pragma unroll
    for (int nt = 0; nt < 4; ++nt) {
        v8f z = {0.f, 0.f, 0.f, 0.f, 0.f, 0.f, 0.f, 0.f};
        acc2[nt] = z;
    }
    const unsigned short* hrow = sAreg + (size_t)m * H_STRIDE + hi * 16;
#pragma unroll
    for (int kt = 0; kt < 8; ++kt) {
        Frag a;
        const unsigned short* ap = hrow + kt * 32;
        a.q[0] = *(const uint4*)(ap);
        a.q[1] = *(const uint4*)(ap + 8);
#pragma unroll
        for (int nt = 0; nt < 4; ++nt) {
            const unsigned short* bp = sW2 +
                    ((size_t)(kt * 4 + nt) * 512 + lane * 16);
            Frag b;
            b.q[0] = *(const uint4*)(bp);
            b.q[1] = *(const uint4*)(bp + 8);
            acc2[nt] = __builtin_amdgcn_wmma_f32_16x16x32_bf16(
                    false, a.v, false, b.v, (short)0, acc2[nt], false, false);
        }
    }

    // ---- Epilogue: bias + residual + LayerNorm(64) + affine ----------------
    float vals[4][8];
    float s[8], ss[8];
#pragma unroll
    for (int r = 0; r < 8; ++r) { s[r] = 0.f; ss[r] = 0.f; }
#pragma unroll
    for (int nt = 0; nt < 4; ++nt) {
        int col = nt * 16 + m;
        float bv = b2[col];
#pragma unroll
        for (int r = 0; r < 8; ++r) {
            int ge = tilebase + r + hi * 8;
            float e = (ge < E) ? edge_attr[(size_t)ge * EDGE_DIM + col] : 0.f;
            float v = acc2[nt][r] + bv + e;
            vals[nt][r] = v;
            s[r]  += v;
            ss[r] += v * v;
        }
    }
    // Reduce across the 16 lanes of each half-wave (xor masks < 16).
#pragma unroll
    for (int mask = 1; mask < 16; mask <<= 1) {
#pragma unroll
        for (int r = 0; r < 8; ++r) {
            s[r]  += __shfl_xor(s[r],  mask, 32);
            ss[r] += __shfl_xor(ss[r], mask, 32);
        }
    }
    float mu[8], rstd[8];
#pragma unroll
    for (int r = 0; r < 8; ++r) {
        mu[r] = s[r] * (1.f / 64.f);
        float var = ss[r] * (1.f / 64.f) - mu[r] * mu[r];
        rstd[r] = __frsqrt_rn(var + 1e-5f);
    }
#pragma unroll
    for (int nt = 0; nt < 4; ++nt) {
        int col = nt * 16 + m;
        float gw = ln_w[col], gb = ln_b[col];
#pragma unroll
        for (int r = 0; r < 8; ++r) {
            int ge = tilebase + r + hi * 8;
            if (ge < E)
                out[(size_t)ge * EDGE_DIM + col] =
                        (vals[nt][r] - mu[r]) * rstd[r] * gw + gb;
        }
    }
}

// ---------------------------------------------------------------------------
extern "C" void kernel_launch(void* const* d_in, const int* in_sizes, int n_in,
                              void* d_out, int out_size, void* d_ws, size_t ws_size,
                              hipStream_t stream) {
    const float*     x    = (const float*)d_in[0];
    const long long* eidx = (const long long*)d_in[1];   // int64 per reference
    const float*     ea   = (const float*)d_in[2];
    const float*     w1   = (const float*)d_in[3];
    const float*     b1   = (const float*)d_in[4];
    const float*     w2   = (const float*)d_in[5];
    const float*     b2   = (const float*)d_in[6];
    const float*     lnw  = (const float*)d_in[7];
    const float*     lnb  = (const float*)d_in[8];
    unsigned short*  wf   = (unsigned short*)d_ws;       // 192KB bf16 fragments
    float*           out  = (float*)d_out;

    const int E = in_sizes[2] / EDGE_DIM;

    // Repack weights to bf16 WMMA fragments (deterministic, every call).
    pack_weights_kernel<<<(192 * 32 + 255) / 256, 256, 0, stream>>>(w1, w2, wf);

    const int nblocks = (E + 127) / 128;
    const size_t smem = (size_t)LDS_TOTAL_US * sizeof(unsigned short); // ~275KB
    edge_update_kernel<<<nblocks, 256, smem, stream>>>(
            x, eidx, ea, b1, b2, lnw, lnb, wf, out, E);
}
